// T5Attention_22290880266242
// MI455X (gfx1250) — matrix operand order repaired and checked by reading, compiled, not run on hardware
//
#include <hip/hip_runtime.h>

typedef __attribute__((ext_vector_type(16))) __bf16 v16bf;
typedef __attribute__((ext_vector_type(8)))  float  v8f;

union BF16Frag { v16bf v; uint4 q[2]; };

#define S_LEN  2048
#define DMODEL 1024
#define NHEADS 16
#define DHEAD  64
#define BATCH  2

// LDS byte offset of a __shared__ object: flat-LDS aperture keeps the
// wave-relative LDS address in addr[31:0].
__device__ __forceinline__ unsigned lds_off(const void* p) {
  return (unsigned)(size_t)p;
}

// Async global->LDS copy of 64 contiguous bytes (4 x b128).
// Instruction offset applies to BOTH the LDS and the global address.
__device__ __forceinline__ void async_copy64(unsigned lds, const void* gptr) {
  unsigned long long g = (unsigned long long)gptr;
  asm volatile("global_load_async_to_lds_b128 %0, %1, off"           :: "v"(lds), "v"(g) : "memory");
  asm volatile("global_load_async_to_lds_b128 %0, %1, off offset:16" :: "v"(lds), "v"(g) : "memory");
  asm volatile("global_load_async_to_lds_b128 %0, %1, off offset:32" :: "v"(lds), "v"(g) : "memory");
  asm volatile("global_load_async_to_lds_b128 %0, %1, off offset:48" :: "v"(lds), "v"(g) : "memory");
}

// Async global->LDS copy of 32 contiguous bytes (2 x b128).
__device__ __forceinline__ void async_copy32(unsigned lds, const void* gptr) {
  unsigned long long g = (unsigned long long)gptr;
  asm volatile("global_load_async_to_lds_b128 %0, %1, off"           :: "v"(lds), "v"(g) : "memory");
  asm volatile("global_load_async_to_lds_b128 %0, %1, off offset:16" :: "v"(lds), "v"(g) : "memory");
}

// ---------------------------------------------------------------- bucketing
__device__ __forceinline__ int t5_bucket(int rel) {
  // bidirectional T5 bucketing: 32 buckets, max_distance 128
  int bucket = (rel > 0) ? 16 : 0;
  float r = fabsf((float)rel);
  if (r < 8.0f) {
    bucket += (int)r;
  } else {
    // 8 + ln(r/8) / ln(128/8) * 8  ;  8/ln(16) = 2.885390082
    float large = 8.0f + __logf(r * 0.125f) * 2.885390082f;
    large = fminf(large, 15.0f);
    bucket += (int)large;
  }
  return bucket;
}

// ---------------------------------------------------------------- converts
__global__ void f32_to_bf16_kernel(const float* __restrict__ in,
                                   __bf16* __restrict__ out, int n) {
  int i = blockIdx.x * blockDim.x + threadIdx.x;
  if (i < n) out[i] = (__bf16)in[i];
}

__global__ __launch_bounds__(256) void transpose_to_bf16_kernel(
    const float* __restrict__ W, __bf16* __restrict__ Wt, int Kdim, int N) {
  __shared__ float tile[32][33];
  int n0 = blockIdx.x * 32, k0 = blockIdx.y * 32;
  int tx = threadIdx.x & 31, ty = threadIdx.x >> 5;
  for (int i = ty; i < 32; i += 8)
    tile[i][tx] = W[(size_t)(k0 + i) * N + n0 + tx];
  __syncthreads();
  for (int i = ty; i < 32; i += 8)
    Wt[(size_t)(n0 + i) * Kdim + k0 + tx] = (__bf16)tile[tx][i];
}

// ---------------------------------------------------------------- GEMM
// C[M][N] = A[M][K] * B[K][N], with B supplied transposed as Bt[N][K].
// 128x128 block tile, 256 threads (8 waves), each wave: 32x64 (2x4 WMMA frags).
// Double-buffered LDS fed by GLOBAL_LOAD_ASYNC_TO_LDS_B128 (ASYNCcnt).
__global__ __launch_bounds__(256) void gemm_bf16_128x128(
    const __bf16* __restrict__ A, const __bf16* __restrict__ Bt,
    float* __restrict__ Cf, __bf16* __restrict__ Cb,
    int M, int N, int Kdim) {
  __shared__ __align__(16) __bf16 As[2][128][40];
  __shared__ __align__(16) __bf16 Bs[2][128][40];

  const int tid  = threadIdx.x;
  const int lane = tid & 31;
  const int wave = tid >> 5;
  const int wM = wave & 3;   // 4 waves along M
  const int wN = wave >> 2;  // 2 waves along N
  const int m0 = blockIdx.y * 128;
  const int n0 = blockIdx.x * 128;

  const v8f vzero = {0.f, 0.f, 0.f, 0.f, 0.f, 0.f, 0.f, 0.f};
  v8f acc[2][4];
  for (int i = 0; i < 2; i++)
    for (int j = 0; j < 4; j++) acc[i][j] = vzero;

  const int ldRow  = tid >> 1;        // 0..127
  const int ldHalf = (tid & 1) * 16;  // element offset within 32-wide K slab

  const __bf16* gA = A  + (size_t)(m0 + ldRow) * Kdim + ldHalf;
  const __bf16* gB = Bt + (size_t)(n0 + ldRow) * Kdim + ldHalf;
  unsigned ldsA[2], ldsB[2];
  for (int s = 0; s < 2; s++) {
    ldsA[s] = lds_off(&As[s][ldRow][ldHalf]);
    ldsB[s] = lds_off(&Bs[s][ldRow][ldHalf]);
  }

  // prime stage 0
  async_copy32(ldsA[0], gA);
  async_copy32(ldsB[0], gB);

  const int T = Kdim >> 5;
  for (int t = 0; t < T; t++) {
    const int cur = t & 1;
    if (t + 1 < T) {
      const int kk = (t + 1) << 5;
      async_copy32(ldsA[cur ^ 1], gA + kk);
      async_copy32(ldsB[cur ^ 1], gB + kk);
      // 4 loads of stage t+1 may stay in flight; stage t's 4 are done.
      asm volatile("s_wait_asynccnt 0x4" ::: "memory");
    } else {
      asm volatile("s_wait_asynccnt 0x0" ::: "memory");
    }
    __syncthreads();

    const int koffA = (lane >> 4) * 8;   // A frag: K halves {0..7,16..23}/{8..15,24..31}
    const int koffB = (lane >> 4) * 8;   // B frag: K halves {0..7,16..23}/{8..15,24..31}
    BF16Frag af[2], bf[4];
    for (int mi = 0; mi < 2; mi++) {
      const __bf16* p = &As[cur][wM * 32 + mi * 16 + (lane & 15)][0];
      af[mi].q[0] = *(const uint4*)(p + koffA);
      af[mi].q[1] = *(const uint4*)(p + koffA + 16);
    }
    for (int ni = 0; ni < 4; ni++) {
      const __bf16* p = &Bs[cur][wN * 64 + ni * 16 + (lane & 15)][0];
      bf[ni].q[0] = *(const uint4*)(p + koffB);
      bf[ni].q[1] = *(const uint4*)(p + koffB + 16);
    }
    for (int mi = 0; mi < 2; mi++)
      for (int ni = 0; ni < 4; ni++)
        acc[mi][ni] = __builtin_amdgcn_wmma_f32_16x16x32_bf16(
            false, af[mi].v, false, bf[ni].v, (short)0, acc[mi][ni], false, false);
    __syncthreads();
  }

  // epilogue: C layout — lane n = l&15, rows r + (l>>4)*8
  const int rowHi = (lane >> 4) * 8;
  for (int mi = 0; mi < 2; mi++) {
    for (int ni = 0; ni < 4; ni++) {
      const int n = n0 + wN * 64 + ni * 16 + (lane & 15);
      for (int r = 0; r < 8; r++) {
        const int m = m0 + wM * 32 + mi * 16 + rowHi + r;
        float v = acc[mi][ni][r];
        if (Cf) Cf[(size_t)m * N + n] = v;
        if (Cb) Cb[(size_t)m * N + n] = (__bf16)v;
      }
    }
  }
}

// ---------------------------------------------------------------- attention
// Flash-style online softmax. Block: 64 q-rows, 128 threads = 4 waves,
// wave owns 16 q-rows. Q/K/V are bf16 [B][S][H][Dh].
__global__ __launch_bounds__(128) void t5_flash_attn(
    const __bf16* __restrict__ Q, const __bf16* __restrict__ K,
    const __bf16* __restrict__ V, const float* __restrict__ table,
    __bf16* __restrict__ AO) {
  __shared__ __align__(16) __bf16 Ks[64][72];    // [key][d], row-major
  __shared__ __align__(16) __bf16 Vts[64][72];   // [d][key], transposed
  __shared__ __align__(16) __bf16 Ps[4][16][64]; // per-wave P tile

  const int tid  = threadIdx.x;
  const int lane = tid & 31;
  const int wave = tid >> 5;
  const int q0 = blockIdx.x * 64;
  const int h  = blockIdx.y;
  const int b  = blockIdx.z;

  // Q fragments in A layout: row = lane&15, K-halves by lane>>4
  BF16Frag qa[2];
  {
    const int q    = q0 + wave * 16 + (lane & 15);
    const int koff = (lane >> 4) * 8;
    const __bf16* qp = Q + ((size_t)(b * S_LEN + q) * NHEADS + h) * DHEAD;
    for (int c = 0; c < 2; c++) {
      qa[c].q[0] = *(const uint4*)(qp + c * 32 + koff);
      qa[c].q[1] = *(const uint4*)(qp + c * 32 + koff + 16);
    }
  }

  const v8f vzero = {0.f, 0.f, 0.f, 0.f, 0.f, 0.f, 0.f, 0.f};
  v8f o[4];
  for (int di = 0; di < 4; di++) o[di] = vzero;
  float mrow[8], lrow[8];
  for (int r = 0; r < 8; r++) { mrow[r] = -1e30f; lrow[r] = 0.f; }

  const int ldRow  = tid >> 1;  // 0..63
  const int ldHalf = tid & 1;
  const unsigned ldsK = lds_off(&Ks[ldRow][ldHalf * 32]);

  for (int k0 = 0; k0 < S_LEN; k0 += 64) {
    __syncthreads();
    {
      const size_t gbase = ((size_t)(b * S_LEN + k0 + ldRow) * NHEADS + h) * DHEAD + ldHalf * 32;
      // K tile: async straight into LDS
      async_copy64(ldsK, K + gbase);
      // V tile: sync load + transposed scatter into LDS
      uint4 u[4];
      const uint4* gv = (const uint4*)(V + gbase);
      u[0] = gv[0]; u[1] = gv[1]; u[2] = gv[2]; u[3] = gv[3];
      const __bf16* tv = (const __bf16*)u;
      for (int j = 0; j < 32; j++) Vts[ldHalf * 32 + j][ldRow] = tv[j];
    }
    asm volatile("s_wait_asynccnt 0x0" ::: "memory");
    __syncthreads();

    // scores: S = Q * K^T  (+ bias)
    const int koffB = (lane >> 4) * 8;
    v8f sfr[4];
    for (int ni = 0; ni < 4; ni++) {
      v8f s = vzero;
      for (int c = 0; c < 2; c++) {
        BF16Frag bf;
        const __bf16* p = &Ks[ni * 16 + (lane & 15)][c * 32 + koffB];
        bf.q[0] = *(const uint4*)p;
        bf.q[1] = *(const uint4*)(p + 16);
        s = __builtin_amdgcn_wmma_f32_16x16x32_bf16(
            false, qa[c].v, false, bf.v, (short)0, s, false, false);
      }
      const int key = k0 + ni * 16 + (lane & 15);
      for (int r = 0; r < 8; r++) {
        const int q = q0 + wave * 16 + (lane >> 4) * 8 + r;
        s[r] += table[t5_bucket(key - q) * NHEADS + h];
      }
      sfr[ni] = s;
    }

    // online softmax (row stats reduced across the 16-lane half-wave)
    for (int r = 0; r < 8; r++) {
      float mx = fmaxf(fmaxf(sfr[0][r], sfr[1][r]), fmaxf(sfr[2][r], sfr[3][r]));
      for (int off = 1; off < 16; off <<= 1) mx = fmaxf(mx, __shfl_xor(mx, off, 16));
      float mnew  = fmaxf(mrow[r], mx);
      float scale = __expf(mrow[r] - mnew);
      mrow[r] = mnew;
      float psum = 0.f;
      for (int ni = 0; ni < 4; ni++) {
        float p = __expf(sfr[ni][r] - mnew);
        sfr[ni][r] = p;
        psum += p;
      }
      for (int off = 1; off < 16; off <<= 1) psum += __shfl_xor(psum, off, 16);
      lrow[r] = lrow[r] * scale + psum;
      for (int di = 0; di < 4; di++) o[di][r] *= scale;
    }

    // P (C layout) -> LDS -> A layout
    for (int ni = 0; ni < 4; ni++)
      for (int r = 0; r < 8; r++)
        Ps[wave][(lane >> 4) * 8 + r][ni * 16 + (lane & 15)] = (__bf16)sfr[ni][r];
    __syncthreads();

    // O += P * V
    const int koffA = (lane >> 4) * 8;
    for (int c = 0; c < 2; c++) {
      BF16Frag pa;
      const __bf16* pp = &Ps[wave][lane & 15][c * 32 + koffA];
      pa.q[0] = *(const uint4*)pp;
      pa.q[1] = *(const uint4*)(pp + 16);
      for (int di = 0; di < 4; di++) {
        BF16Frag vb;
        const __bf16* vp = &Vts[di * 16 + (lane & 15)][c * 32 + koffB];
        vb.q[0] = *(const uint4*)vp;
        vb.q[1] = *(const uint4*)(vp + 16);
        o[di] = __builtin_amdgcn_wmma_f32_16x16x32_bf16(
            false, pa.v, false, vb.v, (short)0, o[di], false, false);
      }
    }
  }

  // normalize and store bf16 attention output
  for (int r = 0; r < 8; r++) lrow[r] = 1.0f / lrow[r];
  for (int di = 0; di < 4; di++) {
    const int d = di * 16 + (lane & 15);
    for (int r = 0; r < 8; r++) {
      const int q = q0 + wave * 16 + (lane >> 4) * 8 + r;
      AO[((size_t)(b * S_LEN + q) * NHEADS + h) * DHEAD + d] = (__bf16)(o[di][r] * lrow[r]);
    }
  }
}

// ---------------------------------------------------------------- bias output
// Writes bias[S][S][H] (256 MB): pure bandwidth, one (q,k) pair per thread.
__global__ __launch_bounds__(256) void write_bias_kernel(
    const float* __restrict__ table, float* __restrict__ out) {
  size_t idx = (size_t)blockIdx.x * blockDim.x + threadIdx.x;  // < S*S
  int q = (int)(idx >> 11);
  int k = (int)(idx & (S_LEN - 1));
  int bucket = t5_bucket(k - q);
  const float4* src = (const float4*)(table + bucket * NHEADS);
  float4* dst = (float4*)(out + idx * NHEADS);
  dst[0] = src[0]; dst[1] = src[1]; dst[2] = src[2]; dst[3] = src[3];
}

// ---------------------------------------------------------------- launch
extern "C" void kernel_launch(void* const* d_in, const int* in_sizes, int n_in,
                              void* d_out, int out_size, void* d_ws, size_t ws_size,
                              hipStream_t stream) {
  const float* x  = (const float*)d_in[0];
  const float* Wq = (const float*)d_in[1];
  const float* Wk = (const float*)d_in[2];
  const float* Wv = (const float*)d_in[3];
  const float* Wo = (const float*)d_in[4];
  const float* tb = (const float*)d_in[5];

  char* ws = (char*)d_ws;
  __bf16* xb  = (__bf16*)(ws);                         //  8 MB
  __bf16* WqT = (__bf16*)(ws + ( 8ull << 20));         //  2 MB
  __bf16* WkT = (__bf16*)(ws + (10ull << 20));         //  2 MB
  __bf16* WvT = (__bf16*)(ws + (12ull << 20));         //  2 MB
  __bf16* WoT = (__bf16*)(ws + (14ull << 20));         //  2 MB
  __bf16* Qb  = (__bf16*)(ws + (16ull << 20));         //  8 MB
  __bf16* Kb  = (__bf16*)(ws + (24ull << 20));         //  8 MB
  __bf16* Vb  = (__bf16*)(ws + (32ull << 20));         //  8 MB
  __bf16* AOb = (__bf16*)(ws + (40ull << 20));         //  8 MB (48 MB total)

  const int M = BATCH * S_LEN;  // 4096
  const int N = DMODEL, Kd = DMODEL;

  f32_to_bf16_kernel<<<(M * Kd + 255) / 256, 256, 0, stream>>>(x, xb, M * Kd);

  dim3 tg(DMODEL / 32, DMODEL / 32);
  transpose_to_bf16_kernel<<<tg, 256, 0, stream>>>(Wq, WqT, Kd, N);
  transpose_to_bf16_kernel<<<tg, 256, 0, stream>>>(Wk, WkT, Kd, N);
  transpose_to_bf16_kernel<<<tg, 256, 0, stream>>>(Wv, WvT, Kd, N);
  transpose_to_bf16_kernel<<<tg, 256, 0, stream>>>(Wo, WoT, Kd, N);

  dim3 gg(N / 128, M / 128);
  gemm_bf16_128x128<<<gg, 256, 0, stream>>>(xb, WqT, nullptr, Qb, M, N, Kd);
  gemm_bf16_128x128<<<gg, 256, 0, stream>>>(xb, WkT, nullptr, Kb, M, N, Kd);
  gemm_bf16_128x128<<<gg, 256, 0, stream>>>(xb, WvT, nullptr, Vb, M, N, Kd);

  dim3 ag(S_LEN / 64, NHEADS, BATCH);
  t5_flash_attn<<<ag, 128, 0, stream>>>(Qb, Kb, Vb, tb, AOb);

  gemm_bf16_128x128<<<gg, 256, 0, stream>>>(AOb, WoT, (float*)d_out, nullptr, M, N, Kd);

  const size_t biasPairs = (size_t)S_LEN * S_LEN;  // 4M threads, 16 floats each
  write_bias_kernel<<<(unsigned)(biasPairs / 256), 256, 0, stream>>>(
      tb, (float*)d_out + (size_t)M * DMODEL);
}
